// EncoderLayer_8160437862705
// MI455X (gfx1250) — compile-verified, matrix operand order checked by reading, not run on hardware
//
#include <hip/hip_runtime.h>
#include <cstddef>
#include <cstdint>

// ---------------------------------------------------------------------------
// Transformer encoder layer for MI455X (gfx1250), bf16 WMMA + f32 accumulate,
// async global->LDS double-buffered GEMM pipeline.
// B=2, S=2048, D=1024, H=16, DK=64, DFF=4096.
// ---------------------------------------------------------------------------

typedef __bf16 bf16;
typedef bf16  v16bf __attribute__((ext_vector_type(16)));
typedef bf16  v8bf  __attribute__((ext_vector_type(8)));
typedef float v8f   __attribute__((ext_vector_type(8)));

#define BM 128
#define BN 128
#define BK 64
#define LDT 72   // BK + 8 halfs pad: row stride 144B = 9*16B (b128-aligned, spreads banks)

#if defined(__AMDGCN__) && __has_builtin(__builtin_amdgcn_global_load_async_to_lds_b128)
#define USE_ASYNC_LDS 1
#else
#define USE_ASYNC_LDS 0
#endif

#if USE_ASYNC_LDS
// Builtin prototype (from hipcc diagnostic):
//   void __builtin_amdgcn_global_load_async_to_lds_b128(
//       int __attribute__((vector_size(16))) __device__ *gaddr,   // AS(1)
//       int __attribute__((vector_size(16))) __shared__ *laddr,   // AS(3)
//       int imm_offset, int cpol)
typedef int v4i_vs __attribute__((vector_size(16)));
typedef __attribute__((address_space(1))) v4i_vs* gv4ip;
typedef __attribute__((address_space(3))) v4i_vs* lv4ip;

// 16B per lane, wave-wide async copy global -> LDS (tracked by ASYNCcnt).
__device__ __forceinline__ void cp_async16(const bf16* g, bf16* l) {
    __builtin_amdgcn_global_load_async_to_lds_b128(
        (gv4ip)(uintptr_t)g,
        (lv4ip)(uint32_t)(uintptr_t)l,
        0, 0);
}
__device__ __forceinline__ void wait_async_le8() {
#if __has_builtin(__builtin_amdgcn_s_wait_asynccnt)
    __builtin_amdgcn_s_wait_asynccnt(8);
#else
    asm volatile("s_wait_asynccnt 0x8" ::: "memory");
#endif
}
__device__ __forceinline__ void wait_async_0() {
#if __has_builtin(__builtin_amdgcn_s_wait_asynccnt)
    __builtin_amdgcn_s_wait_asynccnt(0);
#else
    asm volatile("s_wait_asynccnt 0x0" ::: "memory");
#endif
}
#endif

// Load one 16x32 bf16 A/B fragment from an LDS tile (row-major, stride LDT).
// ISA layout (CDNA5 ISA): lane%16 = row, halves 0-7 = K {0..7 | 8..15},
// halves 8-15 = K {16..23 | 24..31} depending on lane<16 / lane>=16.
__device__ __forceinline__ v16bf frag_ld(const bf16* tile, int row, int lane) {
    int r  = row + (lane & 15);
    int kh = (lane & 16) ? 8 : 0;
    const bf16* p = tile + r * LDT + kh;
    v8bf lo = *(const v8bf*)p;
    v8bf hi = *(const v8bf*)(p + 16);
    v16bf f;
#pragma unroll
    for (int i = 0; i < 8; ++i) { f[i] = lo[i]; f[i + 8] = hi[i]; }
    return f;
}

// C = alpha * (A[MxK] @ BT[NxK]^T) (+bias) (relu); writes fp32 and/or bf16.
// M must be a multiple of BM; N,K arbitrary multiples of 64 (rows clamped, outputs guarded).
template <bool HAS_BIAS, bool DO_RELU>
__global__ __launch_bounds__(256) void gemm_bf16(
    const bf16* __restrict__ A, int lda,
    const bf16* __restrict__ BT, int ldb,
    float* __restrict__ Cf, int ldcf,
    bf16* __restrict__ Cb, int ldcb,
    const float* __restrict__ bias,
    float alpha, int M, int N, int K)
{
    __shared__ bf16 As[2][BM * LDT];
    __shared__ bf16 Bs[2][BN * LDT];

    const int tid  = threadIdx.x;
    const int lane = tid & 31;
    const int w    = tid >> 5;   // 8 waves
    const int wr   = w >> 1;     // 0..3 -> M offset wr*32
    const int wc   = w & 1;      // 0..1 -> N offset wc*64
    const int bm   = blockIdx.y * BM;
    const int bn   = blockIdx.x * BN;

    v8f acc[2][4];
#pragma unroll
    for (int i = 0; i < 2; ++i)
#pragma unroll
        for (int j = 0; j < 4; ++j) acc[i][j] = {};

    // Stage BMxBK A tile and BNxBK BT tile into LDS buffer `buf`.
    // 1024 chunks of 8 bf16 (16B) each per tile; 8 async ops per wave per stage.
    auto stage = [&](int buf, int kb) {
#pragma unroll
        for (int it = 0; it < 4; ++it) {
            int chunk = it * 256 + tid;
            int r = chunk >> 3;
            int c = (chunk & 7) * 8;
            int gm = bm + r; if (gm >= M) gm = M - 1;   // clamp: outputs are guarded
            int gn = bn + r; if (gn >= N) gn = N - 1;
#if USE_ASYNC_LDS
            cp_async16(A  + (size_t)gm * lda + kb + c, &As[buf][r * LDT + c]);
            cp_async16(BT + (size_t)gn * ldb + kb + c, &Bs[buf][r * LDT + c]);
#else
            *(v8bf*)(&As[buf][r * LDT + c]) = *(const v8bf*)(A  + (size_t)gm * lda + kb + c);
            *(v8bf*)(&Bs[buf][r * LDT + c]) = *(const v8bf*)(BT + (size_t)gn * ldb + kb + c);
#endif
        }
    };

    const int nk = K / BK;
    stage(0, 0);

    for (int kt = 0; kt < nk; ++kt) {
        const int cur = kt & 1;
        const bool have_next = (kt + 1 < nk);
        if (have_next) stage(cur ^ 1, (kt + 1) * BK);
#if USE_ASYNC_LDS
        // Oldest 8 async ops are the copies for buffer `cur`; leave next stage in flight.
        if (have_next) wait_async_le8(); else wait_async_0();
#endif
        __syncthreads();

#pragma unroll
        for (int ks = 0; ks < 2; ++ks) {
            v16bf afrag[2], bfrag[4];
#pragma unroll
            for (int i = 0; i < 2; ++i)
                afrag[i] = frag_ld(&As[cur][ks * 32], wr * 32 + i * 16, lane);
#pragma unroll
            for (int j = 0; j < 4; ++j)
                bfrag[j] = frag_ld(&Bs[cur][ks * 32], wc * 64 + j * 16, lane);
#pragma unroll
            for (int i = 0; i < 2; ++i)
#pragma unroll
                for (int j = 0; j < 4; ++j)
                    acc[i][j] = __builtin_amdgcn_wmma_f32_16x16x32_bf16(
                        false, afrag[i], false, bfrag[j],
                        (short)0, acc[i][j], false, false);
        }
        __syncthreads();   // all waves done reading `cur` before it is restaged
    }

    // Epilogue. C/D layout: VGPR r, lanes<16 -> M=r, lanes>=16 -> M=r+8, N=lane%16.
#pragma unroll
    for (int i = 0; i < 2; ++i) {
#pragma unroll
        for (int j = 0; j < 4; ++j) {
            int mbase = bm + wr * 32 + i * 16 + ((lane & 16) ? 8 : 0);
            int n     = bn + wc * 64 + j * 16 + (lane & 15);
            if (n < N) {
                float bv = HAS_BIAS ? bias[n] : 0.0f;
#pragma unroll
                for (int r = 0; r < 8; ++r) {
                    int m = mbase + r;
                    if (m < M) {
                        float v = acc[i][j][r] * alpha;
                        if (HAS_BIAS) v += bv;
                        if (DO_RELU) v = fmaxf(v, 0.0f);
                        if (Cf) Cf[(size_t)m * ldcf + n] = v;
                        if (Cb) Cb[(size_t)m * ldcb + n] = (bf16)v;
                    }
                }
            }
        }
    }
}

// Row softmax over 2048 columns, then multiply by query-row keep mask.
__global__ __launch_bounds__(256) void softmax_mask(
    const float* __restrict__ S, bf16* __restrict__ P,
    const int* __restrict__ mask)
{
    __shared__ float red[256];
    int row = blockIdx.x;
    int t   = threadIdx.x;
    const float* sr = S + (size_t)row * 2048;
    bf16*        pr = P + (size_t)row * 2048;

    float v[8];
    float m = -3.0e38f;
#pragma unroll
    for (int i = 0; i < 8; ++i) { v[i] = sr[t + i * 256]; m = fmaxf(m, v[i]); }
    red[t] = m; __syncthreads();
    for (int s = 128; s > 0; s >>= 1) { if (t < s) red[t] = fmaxf(red[t], red[t + s]); __syncthreads(); }
    m = red[0]; __syncthreads();

    float sum = 0.0f;
#pragma unroll
    for (int i = 0; i < 8; ++i) { v[i] = __expf(v[i] - m); sum += v[i]; }
    red[t] = sum; __syncthreads();
    for (int s = 128; s > 0; s >>= 1) { if (t < s) red[t] += red[t + s]; __syncthreads(); }

    float inv = (mask[row] != 0) ? (1.0f / red[0]) : 0.0f;
#pragma unroll
    for (int i = 0; i < 8; ++i) pr[t + i * 256] = (bf16)(v[i] * inv);
}

// out = x + LayerNorm(y)*gamma+beta over D=1024; optional bf16 copy.
__global__ __launch_bounds__(256) void residual_ln(
    const float* __restrict__ X, const float* __restrict__ Y,
    const float* __restrict__ gamma, const float* __restrict__ beta,
    float* __restrict__ Of, bf16* __restrict__ Ob)
{
    __shared__ float red[256];
    int row = blockIdx.x;
    int t   = threadIdx.x;
    const float* yr = Y + (size_t)row * 1024;
    const float* xr = X + (size_t)row * 1024;

    float v[4];
    float s = 0.0f;
#pragma unroll
    for (int i = 0; i < 4; ++i) { v[i] = yr[t + i * 256]; s += v[i]; }
    red[t] = s; __syncthreads();
    for (int k = 128; k > 0; k >>= 1) { if (t < k) red[t] += red[t + k]; __syncthreads(); }
    float mu = red[0] * (1.0f / 1024.0f); __syncthreads();

    float sq = 0.0f;
#pragma unroll
    for (int i = 0; i < 4; ++i) { float d = v[i] - mu; sq += d * d; }
    red[t] = sq; __syncthreads();
    for (int k = 128; k > 0; k >>= 1) { if (t < k) red[t] += red[t + k]; __syncthreads(); }
    float rstd = rsqrtf(red[0] * (1.0f / 1024.0f) + 1e-6f);

#pragma unroll
    for (int i = 0; i < 4; ++i) {
        int c = t + i * 256;
        float o = xr[c] + (v[i] - mu) * rstd * gamma[c] + beta[c];
        Of[(size_t)row * 1024 + c] = o;
        if (Ob) Ob[(size_t)row * 1024 + c] = (bf16)o;
    }
}

__global__ void cvt_f32_bf16(const float* __restrict__ in, bf16* __restrict__ out, int n) {
    int i = blockIdx.x * 256 + threadIdx.x;
    if (i < n) out[i] = (bf16)in[i];
}

// W[K][N] fp32 -> Wt[N][K] bf16 (32x32 LDS tile transpose). K,N multiples of 32.
__global__ void transpose_cvt(const float* __restrict__ W, bf16* __restrict__ Wt, int K, int N) {
    __shared__ float tile[32][33];
    int n0 = blockIdx.x * 32, k0 = blockIdx.y * 32;
#pragma unroll
    for (int i = 0; i < 4; ++i) {
        int k = k0 + threadIdx.y + i * 8;
        tile[threadIdx.y + i * 8][threadIdx.x] = W[(size_t)k * N + n0 + threadIdx.x];
    }
    __syncthreads();
#pragma unroll
    for (int i = 0; i < 4; ++i) {
        int n = n0 + threadIdx.y + i * 8;
        Wt[(size_t)n * K + k0 + threadIdx.x] = (bf16)tile[threadIdx.x][threadIdx.y + i * 8];
    }
}

// V bf16 [B*S, D] -> VT bf16 [B,H,DK,S]: VT[((b*H+h)*DK+dk)*S + s] = V[(b*S+s)*D + h*DK+dk]
__global__ void v_transpose(const bf16* __restrict__ V, bf16* __restrict__ VT) {
    int o = blockIdx.x * 256 + threadIdx.x;   // 2*16*64*2048 = 4194304
    int s  = o & 2047;
    int r  = o >> 11;
    int dk = r & 63;
    int hh = (r >> 6) & 15;
    int b  = r >> 10;
    VT[o] = V[(((size_t)b * 2048 + s) * 1024) + hh * 64 + dk];
}

extern "C" void kernel_launch(void* const* d_in, const int* in_sizes, int n_in,
                              void* d_out, int out_size, void* d_ws, size_t ws_size,
                              hipStream_t stream) {
    (void)in_sizes; (void)n_in; (void)out_size; (void)ws_size;

    const float* X    = (const float*)d_in[0];
    const int*   mask = (const int*)d_in[1];
    const float* W_Q  = (const float*)d_in[2];
    const float* W_K  = (const float*)d_in[3];
    const float* W_V  = (const float*)d_in[4];
    const float* W_O  = (const float*)d_in[5];
    const float* W1   = (const float*)d_in[6];
    const float* b1   = (const float*)d_in[7];
    const float* W2   = (const float*)d_in[8];
    const float* b2   = (const float*)d_in[9];
    const float* g1   = (const float*)d_in[10];
    const float* be1  = (const float*)d_in[11];
    const float* g2   = (const float*)d_in[12];
    const float* be2  = (const float*)d_in[13];
    float* out = (float*)d_out;

    char* p = (char*)d_ws;
    auto alloc = [&](size_t bytes) -> void* {
        void* r = (void*)p;
        p += (bytes + 255) & ~(size_t)255;
        return r;
    };
    const size_t MT = 4096;            // B*S tokens
    bf16*  Xbf  = (bf16*)alloc(MT * 1024 * 2);
    bf16*  WQt  = (bf16*)alloc(1024ull * 1024 * 2);
    bf16*  WKt  = (bf16*)alloc(1024ull * 1024 * 2);
    bf16*  WVt  = (bf16*)alloc(1024ull * 1024 * 2);
    bf16*  WOt  = (bf16*)alloc(1024ull * 1024 * 2);
    bf16*  W1t  = (bf16*)alloc(4096ull * 1024 * 2);
    bf16*  W2t  = (bf16*)alloc(1024ull * 4096 * 2);
    bf16*  Qbf  = (bf16*)alloc(MT * 1024 * 2);
    bf16*  Kbf  = (bf16*)alloc(MT * 1024 * 2);
    bf16*  Vbf  = (bf16*)alloc(MT * 1024 * 2);
    bf16*  VT   = (bf16*)alloc(2ull * 16 * 64 * 2048 * 2);
    float* scr  = (float*)alloc(2048ull * 2048 * 4);
    bf16*  Pbf  = (bf16*)alloc(2048ull * 2048 * 2);
    bf16*  Ctx  = (bf16*)alloc(MT * 1024 * 2);
    float* attn = (float*)alloc(MT * 1024 * 4);
    float* x1   = (float*)alloc(MT * 1024 * 4);
    bf16*  x1bf = (bf16*)alloc(MT * 1024 * 2);
    bf16*  hbf  = (bf16*)alloc(MT * 4096 * 2);
    float* ffn  = (float*)alloc(MT * 1024 * 4);

    auto gemm = [&](const bf16* A, int lda, const bf16* BT, int ldb,
                    float* Cf, int ldcf, bf16* Cb, int ldcb,
                    const float* bias, float alpha, int M, int N, int K, bool relu) {
        dim3 g((N + BN - 1) / BN, (M + BM - 1) / BM);
        if (bias && relu)
            gemm_bf16<true, true><<<g, 256, 0, stream>>>(A, lda, BT, ldb, Cf, ldcf, Cb, ldcb, bias, alpha, M, N, K);
        else if (bias)
            gemm_bf16<true, false><<<g, 256, 0, stream>>>(A, lda, BT, ldb, Cf, ldcf, Cb, ldcb, bias, alpha, M, N, K);
        else
            gemm_bf16<false, false><<<g, 256, 0, stream>>>(A, lda, BT, ldb, Cf, ldcf, Cb, ldcb, nullptr, alpha, M, N, K);
    };

    // --- precision conversion / weight transposes ---
    cvt_f32_bf16<<<(4194304 + 255) / 256, 256, 0, stream>>>(X, Xbf, 4194304);
    dim3 tb(32, 8);
    transpose_cvt<<<dim3(32, 32),  tb, 0, stream>>>(W_Q, WQt, 1024, 1024);
    transpose_cvt<<<dim3(32, 32),  tb, 0, stream>>>(W_K, WKt, 1024, 1024);
    transpose_cvt<<<dim3(32, 32),  tb, 0, stream>>>(W_V, WVt, 1024, 1024);
    transpose_cvt<<<dim3(32, 32),  tb, 0, stream>>>(W_O, WOt, 1024, 1024);
    transpose_cvt<<<dim3(128, 32), tb, 0, stream>>>(W1, W1t, 1024, 4096);
    transpose_cvt<<<dim3(32, 128), tb, 0, stream>>>(W2, W2t, 4096, 1024);

    // --- Q, K, V projections ---
    gemm(Xbf, 1024, WQt, 1024, nullptr, 0, Qbf, 1024, nullptr, 1.0f, 4096, 1024, 1024, false);
    gemm(Xbf, 1024, WKt, 1024, nullptr, 0, Kbf, 1024, nullptr, 1.0f, 4096, 1024, 1024, false);
    gemm(Xbf, 1024, WVt, 1024, nullptr, 0, Vbf, 1024, nullptr, 1.0f, 4096, 1024, 1024, false);
    v_transpose<<<16384, 256, 0, stream>>>(Vbf, VT);

    // --- attention per (b, h): scores -> softmax(+query mask) -> P @ V ---
    for (int b = 0; b < 2; ++b) {
        for (int h = 0; h < 16; ++h) {
            const bf16* Qh  = Qbf + (size_t)b * 2048 * 1024 + h * 64;
            const bf16* Kh  = Kbf + (size_t)b * 2048 * 1024 + h * 64;
            const bf16* VTh = VT + ((size_t)(b * 16 + h)) * 64 * 2048;
            gemm(Qh, 1024, Kh, 1024, scr, 2048, nullptr, 0, nullptr, 0.125f, 2048, 2048, 64, false);
            softmax_mask<<<2048, 256, 0, stream>>>(scr, Pbf, mask + b * 2048);
            gemm(Pbf, 2048, VTh, 2048, nullptr, 0,
                 Ctx + (size_t)b * 2048 * 1024 + h * 64, 1024,
                 nullptr, 1.0f, 2048, 64, 2048, false);
        }
    }

    // --- output projection + residual LN ---
    gemm(Ctx, 1024, WOt, 1024, attn, 1024, nullptr, 0, nullptr, 1.0f, 4096, 1024, 1024, false);
    residual_ln<<<4096, 256, 0, stream>>>(X, attn, g1, be1, x1, x1bf);

    // --- FFN ---
    gemm(x1bf, 1024, W1t, 1024, nullptr, 0, hbf, 4096, b1, 1.0f, 4096, 4096, 1024, true);
    gemm(hbf, 4096, W2t, 4096, ffn, 1024, nullptr, 0, b2, 1.0f, 4096, 1024, 4096, false);
    residual_ln<<<4096, 256, 0, stream>>>(x1, ffn, g2, be2, out, nullptr);
}
